// MOAB_46273977647401
// MI455X (gfx1250) — matrix operand order repaired and checked by reading, not compiled, not run on hardware
//
#include <hip/hip_runtime.h>
#include <hip/hip_bf16.h>

// ---------------- problem constants ----------------
#define B_SZ   256
#define N_SZ   256
#define H_SZ   512
#define C_SZ   4
#define K_EL   66049            // 257*257
#define K_MAIN 66048            // 32 * 2064 (WMMA-covered K)
#define K_PAD  66064            // row stride, multiple of 16 elems (32B aligned rows)
#define S_SPL  16               // K-split factor
#define KSTEPS 129              // 2064 / 16 k-steps of 32 per chunk

typedef __attribute__((ext_vector_type(16))) __bf16         v16bf;
typedef __attribute__((ext_vector_type(8)))  float          v8f;
typedef __attribute__((ext_vector_type(8)))  unsigned short u16x8;
typedef __attribute__((ext_vector_type(16))) unsigned short u16x16;

static __device__ __forceinline__ unsigned short f2bf(float f) {
  unsigned u = __float_as_uint(f);
  unsigned r = u + 0x7FFFu + ((u >> 16) & 1u);   // round-to-nearest-even
  return (unsigned short)(r >> 16);
}
static __device__ __forceinline__ float bf2f(unsigned short h) {
  return __uint_as_float(((unsigned)h) << 16);
}
static __device__ __forceinline__ float sigmoidf_(float x) {
  return 1.0f / (1.0f + __expf(-x));
}
static __device__ __forceinline__ float leakyf_(float x) {
  return x >= 0.0f ? x : 0.1f * x;
}
// combine two 16B chunks into a 16-element bf16 WMMA operand
static __device__ __forceinline__ v16bf load_op(const unsigned short* __restrict__ p,
                                                int second_off_elems) {
  u16x8 lo = *(const u16x8*)p;
  u16x8 hi = *(const u16x8*)(p + second_off_elems);
  u16x16 c;
#pragma unroll
  for (int i = 0; i < 8; ++i) { c[i] = lo[i]; c[i + 8] = hi[i]; }
  return __builtin_bit_cast(v16bf, c);
}

// ---------------- kernel 1: fc_w (f32, stride 66049) -> bf16 (stride K_PAD, zero-padded)
__global__ __launch_bounds__(256) void wconv_kernel(const float* __restrict__ fc_w,
                                                    unsigned short* __restrict__ Wbf) {
  size_t idx = (size_t)blockIdx.x * 256 + threadIdx.x;  // one 8-col chunk each
  size_t row = idx / (K_PAD / 8);
  size_t col = (idx % (K_PAD / 8)) * 8;
  if (row >= H_SZ) return;
  u16x8 o;
#pragma unroll
  for (int t = 0; t < 8; ++t) {
    size_t c = col + t;
    o[t] = (c < K_EL) ? f2bf(fc_w[row * (size_t)K_EL + c]) : (unsigned short)0;
  }
  *(u16x8*)(Wbf + row * (size_t)K_PAD + col) = o;
}

// ---------------- kernel 2: build feature matrix F (256 x K_PAD) in bf16
__global__ __launch_bounds__(256) void featgen_kernel(
    const float* __restrict__ x1, const float* __restrict__ x3,
    const float* __restrict__ cw, const float* __restrict__ cb,
    const float* __restrict__ g,  const float* __restrict__ be,
    const float* __restrict__ mu, const float* __restrict__ var,
    unsigned short* __restrict__ Fbf) {
  size_t idx = (size_t)blockIdx.x * 256 + threadIdx.x;
  size_t b   = idx / K_PAD;
  size_t col = idx % K_PAD;
  if (b >= B_SZ) return;
  unsigned short out = 0;
  if (col < K_EL) {
    int i = (int)(col / 257);
    int j = (int)(col % 257);
    float xi = (i > 0) ? x3[i - 1] : 0.0f;
    float b0 = (i > 0) ? xi : 0.0f;
    float b1 = (i > 0) ? xi : 1.0f;
    float xj = (j > 0) ? x1[b * N_SZ + (j - 1)] : 0.0f;
    float a0 = (j > 0) ? xj : 0.0f;
    float a1 = (j > 0) ? xj : 1.0f;

    float x_add = sigmoidf_(b0 + a0);
    float x_sub = sigmoidf_(b0 - a0);
    float x_pro = sigmoidf_(b1 * a1);
    float x_div = leakyf_(sigmoidf_(b1 * __builtin_amdgcn_rcpf(a1 + 1e-10f)));

    float y = cw[0] * x_add + cw[1] * x_sub + cw[2] * x_pro + cw[3] * x_div + cb[0];
    float scale = g[0] * __frsqrt_rn(var[0] + 1e-5f);
    y = (y - mu[0]) * scale + be[0];
    y = leakyf_(y);
    out = f2bf(y);
  }
  Fbf[idx] = out;
}

// ---------------- kernel 3: bf16 WMMA GEMM  Ypart[s] += F(64x...) * W^T(...x32)
__global__ __launch_bounds__(256) void gemm_kernel(
    const unsigned short* __restrict__ Wbf,
    const unsigned short* __restrict__ Fbf,
    float* __restrict__ Ypart) {
  const int lane = threadIdx.x & 31;
  const int wave = threadIdx.x >> 5;
  const int w    = blockIdx.x * 8 + wave;     // 0..1023
  const int s    = w & (S_SPL - 1);           // k-chunk
  const int nb   = (w >> 4) & 15;             // 16 n-blocks of 32
  const int mb   = w >> 8;                    // 4 m-blocks of 64
  const int half = lane >> 4;
  const int lm   = lane & 15;
  const int m0 = mb * 64;
  const int n0 = nb * 32;
  const size_t kbeg = (size_t)s * (KSTEPS * 32);

  v8f acc[4][2];
#pragma unroll
  for (int mi = 0; mi < 4; ++mi)
#pragma unroll
    for (int ni = 0; ni < 2; ++ni)
#pragma unroll
      for (int v = 0; v < 8; ++v) acc[mi][ni][v] = 0.0f;

  const unsigned short* aB[4];
#pragma unroll
  for (int mi = 0; mi < 4; ++mi)
    aB[mi] = Fbf + (size_t)(m0 + mi * 16 + lm) * K_PAD + kbeg + half * 8;
  const unsigned short* bB[2];
#pragma unroll
  for (int ni = 0; ni < 2; ++ni)
    bB[ni] = Wbf + (size_t)(n0 + ni * 16 + lm) * K_PAD + kbeg + half * 8;

  for (int kk = 0; kk < KSTEPS; ++kk) {
    const size_t ko = (size_t)kk * 32;
    v16bf a[4], b[2];
#pragma unroll
    for (int mi = 0; mi < 4; ++mi) a[mi] = load_op(aB[mi] + ko, 16);
#pragma unroll
    for (int ni = 0; ni < 2; ++ni) b[ni] = load_op(bB[ni] + ko, 16);
#pragma unroll
    for (int mi = 0; mi < 4; ++mi)
#pragma unroll
      for (int ni = 0; ni < 2; ++ni)
        acc[mi][ni] = __builtin_amdgcn_wmma_f32_16x16x32_bf16(
            false, a[mi], false, b[ni], (short)0, acc[mi][ni], false, false);
  }

  // exact tail: K column 66048 handled by the last chunk with scalar FMAs
  if (s == S_SPL - 1) {
    float wv[2];
#pragma unroll
    for (int ni = 0; ni < 2; ++ni)
      wv[ni] = bf2f(Wbf[(size_t)(n0 + ni * 16 + lm) * K_PAD + K_MAIN]);
#pragma unroll
    for (int mi = 0; mi < 4; ++mi) {
#pragma unroll
      for (int v = 0; v < 8; ++v) {
        const int row = m0 + mi * 16 + v + half * 8;
        const float fv = bf2f(Fbf[(size_t)row * K_PAD + K_MAIN]);
#pragma unroll
        for (int ni = 0; ni < 2; ++ni) acc[mi][ni][v] += fv * wv[ni];
      }
    }
  }

  float* Y = Ypart + (size_t)s * B_SZ * H_SZ;
#pragma unroll
  for (int mi = 0; mi < 4; ++mi)
#pragma unroll
    for (int ni = 0; ni < 2; ++ni)
#pragma unroll
      for (int v = 0; v < 8; ++v) {
        const int row = m0 + mi * 16 + v + half * 8;   // batch index
        const int col = n0 + ni * 16 + lm;             // hidden index
        Y[(size_t)row * H_SZ + col] = acc[mi][ni][v];
      }
}

// ---------------- kernel 4: reduce K-slices, fc_b + relu, 512->4 output layer
__global__ __launch_bounds__(128) void final_kernel(
    const float* __restrict__ Ypart, const float* __restrict__ fc_b,
    const float* __restrict__ out_w, const float* __restrict__ out_b,
    float* __restrict__ out) {
  const int b = blockIdx.x;
  const int t = threadIdx.x;
  float p[4] = {0.f, 0.f, 0.f, 0.f};
  for (int h = t; h < H_SZ; h += 128) {
    float a = 0.0f;
#pragma unroll
    for (int s = 0; s < S_SPL; ++s)
      a += Ypart[(size_t)s * B_SZ * H_SZ + (size_t)b * H_SZ + h];
    a += fc_b[h];
    a = fmaxf(a, 0.0f);
#pragma unroll
    for (int c = 0; c < 4; ++c) p[c] += a * out_w[c * H_SZ + h];
  }
  __shared__ float red[4][128];
#pragma unroll
  for (int c = 0; c < 4; ++c) red[c][t] = p[c];
  __syncthreads();
  for (int off = 64; off > 0; off >>= 1) {
    if (t < off) {
#pragma unroll
      for (int c = 0; c < 4; ++c) red[c][t] += red[c][t + off];
    }
    __syncthreads();
  }
  if (t < 4) out[b * 4 + t] = red[t][0] + out_b[t];
}

// ---------------- launcher ----------------
extern "C" void kernel_launch(void* const* d_in, const int* in_sizes, int n_in,
                              void* d_out, int out_size, void* d_ws, size_t ws_size,
                              hipStream_t stream) {
  const float* x1      = (const float*)d_in[0];
  const float* x3      = (const float*)d_in[1];
  const float* conv_w  = (const float*)d_in[2];
  const float* conv_b  = (const float*)d_in[3];
  const float* bn_g    = (const float*)d_in[4];
  const float* bn_b    = (const float*)d_in[5];
  const float* bn_m    = (const float*)d_in[6];
  const float* bn_v    = (const float*)d_in[7];
  const float* fc_w    = (const float*)d_in[8];
  const float* fc_b    = (const float*)d_in[9];
  const float* out_w   = (const float*)d_in[10];
  const float* out_b   = (const float*)d_in[11];
  float* logits        = (float*)d_out;

  unsigned short* Wbf = (unsigned short*)d_ws;                       // 512 * K_PAD bf16
  unsigned short* Fbf = Wbf + (size_t)H_SZ * K_PAD;                  // 256 * K_PAD bf16
  float*        Ypart = (float*)(Fbf + (size_t)B_SZ * K_PAD);       // S_SPL * 256 * 512 f32

  // 1) convert fc_w -> padded bf16
  {
    const int total = (int)((size_t)H_SZ * (K_PAD / 8));             // 4,228,096
    wconv_kernel<<<total / 256, 256, 0, stream>>>(fc_w, Wbf);
  }
  // 2) feature matrix
  {
    const int total = (int)((size_t)B_SZ * K_PAD);                   // 16,912,384
    featgen_kernel<<<total / 256, 256, 0, stream>>>(
        x1, x3, conv_w, conv_b, bn_g, bn_b, bn_m, bn_v, Fbf);
  }
  // 3) WMMA GEMM (1024 waves = 128 blocks x 8 waves)
  gemm_kernel<<<128, 256, 0, stream>>>(Wbf, Fbf, Ypart);
  // 4) reduce + epilogue
  final_kernel<<<B_SZ, 128, 0, stream>>>(Ypart, fc_b, out_w, out_b, logits);
}
